// RelativeGlobalAttention_37211596653048
// MI455X (gfx1250) — matrix operand order repaired and checked by reading, not compiled, not run on hardware
//
#include <hip/hip_runtime.h>
#include <hip/hip_bf16.h>
#include <math.h>

// ---------------------------------------------------------------------------
// RelativeGlobalAttention forward for MI455X (gfx1250, wave32, WMMA + TDM).
//   B=4, S=1024, D=1024, H=16, dh=64, causal; skew identity:
//   Srel[s,t] = q[s] . Er[t - s + S - 1]   (t <= s)
// Phase 0: f32 -> f16 pre-convert (x, Wq, Wk, Wv, Er) so GEMM loops are
//          pure b128 + WMMA (no per-tile v_cvt chains).
// Phase 1: q/k = x @ W^T -> f16 [B,H,S,dh]; v -> f16 transposed [B,H,dh,S]
// Phase 2: per 64-row query block: QEr strip (16x1024/wave) via WMMA to LDS,
//          then flash attention over 32-key blocks with K/V tiles streamed by
//          the Tensor Data Mover into double-buffered LDS (s_wait_tensorcnt).
// ---------------------------------------------------------------------------

typedef __attribute__((ext_vector_type(16))) _Float16 v16h;
typedef __attribute__((ext_vector_type(8)))  _Float16 v8h;
typedef __attribute__((ext_vector_type(8)))  float    v8f;
typedef __attribute__((ext_vector_type(4)))  float    v4f;
typedef __attribute__((ext_vector_type(4)))  unsigned int v4u;
typedef __attribute__((ext_vector_type(8)))  int      v8i;
typedef __attribute__((ext_vector_type(4)))  int      v4i;

#define BATCH 4
#define SEQ   1024
#define DMODEL 1024
#define NHEAD 16
#define DHEAD 64

__device__ __forceinline__ int lane_id() { return (int)(threadIdx.x & 31); }

__device__ __forceinline__ v8f wmma_f16(v16h a, v16h b, v8f c) {
  return __builtin_amdgcn_wmma_f32_16x16x32_f16(false, a, false, b, (short)0, c,
                                                false, false);
}

// A-matrix fragment (16x32 f16): row M = lane&15, half h = lane>>4,
// elems 0..7 -> K = 8h+j ; elems 8..15 -> K = 8h+16+j
__device__ __forceinline__ v16h frag_a_f16(const _Float16* base, int ldk) {
  const int lane = lane_id();
  const _Float16* p = base + (lane & 15) * ldk + ((lane >> 4) << 3);
  v8h lo = *(const v8h*)p;
  v8h hi = *(const v8h*)(p + 16);
  v16h f;
#pragma unroll
  for (int j = 0; j < 8; ++j) { f[j] = lo[j]; f[j + 8] = hi[j]; }
  return f;
}

// B-matrix fragment (32x16 f16): col N = lane&15, elems 0..7 -> K = 8h+e ;
// elems 8..15 -> K = 8h+16+e.
// Source is N-major rows (B^T row-major), contiguous along K.
__device__ __forceinline__ v16h frag_b_f16(const _Float16* base, int ldk) {
  const int lane = lane_id();
  const _Float16* p = base + (lane & 15) * ldk + ((lane >> 4) << 3);
  v8h lo = *(const v8h*)p;
  v8h hi = *(const v8h*)(p + 16);
  v16h f;
#pragma unroll
  for (int e = 0; e < 8; ++e) { f[e] = lo[e]; f[e + 8] = hi[e]; }
  return f;
}

// Row reductions across the 16 lanes of each half-wave.
__device__ __forceinline__ float rowmax16(float x) {
#pragma unroll
  for (int m = 8; m >= 1; m >>= 1) x = fmaxf(x, __shfl_xor(x, m, 32));
  return x;
}
__device__ __forceinline__ float rowsum16(float x) {
#pragma unroll
  for (int m = 8; m >= 1; m >>= 1) x += __shfl_xor(x, m, 32);
  return x;
}

// TDM: load a tile_h x tile_w (f16 elements) tile, row stride `stride`
// (elements), from global into LDS at byte offset lds_byte_off.
// D# layout per CDNA5 ISA ch.8 (group0: count/lds/global/type=2; group1:
// data_size=2B, tensor dims = tile dims (tile fully in-bounds), dim0 stride).
// This toolchain's builtin takes 6 args: (g0, g1, g2, g3, g_extra, cpol);
// groups 2/3 + extra are zero (tensor rank <= 2).
__device__ __forceinline__ void tdm_load_2d(unsigned lds_byte_off,
                                            const _Float16* gsrc,
                                            unsigned tile_w, unsigned tile_h,
                                            unsigned stride) {
  unsigned long long ga = (unsigned long long)(size_t)gsrc;
  v4u g0 = {1u,                                   // count=1, user mode
            lds_byte_off,                         // [63:32] lds_addr
            (unsigned)ga,                         // [95:64] global_addr lo
            ((unsigned)(ga >> 32) & 0x01FFFFFFu) | (2u << 30)};  // type=2
  v8i g1 = {(int)(1u << 16),                              // data_size = 2B
            (int)(tile_w << 16),                          // tensor_dim0 lo16
            (int)((tile_w >> 16) | (tile_h << 16)),       // dim0 hi | dim1 lo
            (int)((tile_h >> 16) | (tile_w << 16)),       // dim1 hi | tile_dim0
            (int)tile_h,                                  // tile_dim1
            (int)stride,                                  // dim0_stride lo32
            0, 0};
  v4i z4 = {0, 0, 0, 0};
  v8i z8 = {0, 0, 0, 0, 0, 0, 0, 0};
  __builtin_amdgcn_tensor_load_to_lds(g0, g1, z4, z4, z8, 0);
}

// ---------------------------------------------------------------------------
// Phase 0: f32 -> f16 convert (8 elements per thread, b128 in / b128 out)
// ---------------------------------------------------------------------------
__global__ void __launch_bounds__(256)
cvt_f16_kernel(const float* __restrict__ src, _Float16* __restrict__ dst,
               int n8) {
  const int i = blockIdx.x * 256 + threadIdx.x;
  if (i < n8) {
    v4f a = ((const v4f*)src)[2 * i];
    v4f b = ((const v4f*)src)[2 * i + 1];
    v8h o;
#pragma unroll
    for (int j = 0; j < 4; ++j) {
      o[j] = (_Float16)a[j];
      o[j + 4] = (_Float16)b[j];
    }
    ((v8h*)dst)[i] = o;
  }
}

// ---------------------------------------------------------------------------
// Phase 1: projections. grid (16,16,3); block 256 (8 waves).
// Wave: 32 rows x 64 cols (2 A-frags, 8 C-tiles) per k-step of 32.
// ---------------------------------------------------------------------------
__global__ void __launch_bounds__(256)
qkv_proj_kernel(const _Float16* __restrict__ x16,
                const _Float16* __restrict__ Wq16,
                const _Float16* __restrict__ Wk16,
                const _Float16* __restrict__ Wv16,
                _Float16* __restrict__ q16,
                _Float16* __restrict__ k16,
                _Float16* __restrict__ vT16) {
  const int wave = threadIdx.x >> 5;
  const int lane = lane_id();
  const int half = lane >> 4;
  const int nIdx = lane & 15;
  const int which = blockIdx.z;
  const _Float16* W = (which == 0) ? Wq16 : (which == 1) ? Wk16 : Wv16;

  const int m0 = blockIdx.x * 256 + wave * 32;  // row in [B*S]
  const int n0 = blockIdx.y * 64;               // col in [D]

  v8f acc[2][4] = {};
  for (int kk = 0; kk < DMODEL; kk += 32) {
    const v16h a0 = frag_a_f16(x16 + m0 * DMODEL + kk, DMODEL);
    const v16h a1 = frag_a_f16(x16 + (m0 + 16) * DMODEL + kk, DMODEL);
#pragma unroll
    for (int nt = 0; nt < 4; ++nt) {
      const v16h bf = frag_b_f16(W + (n0 + nt * 16) * DMODEL + kk, DMODEL);
      acc[0][nt] = wmma_f16(a0, bf, acc[0][nt]);
      acc[1][nt] = wmma_f16(a1, bf, acc[1][nt]);
    }
  }

#pragma unroll
  for (int mi = 0; mi < 2; ++mi) {
#pragma unroll
    for (int nt = 0; nt < 4; ++nt) {
#pragma unroll
      for (int i = 0; i < 8; ++i) {
        const int m = m0 + mi * 16 + i + (half << 3);
        const int b = m >> 10, s = m & (SEQ - 1);
        const int n = n0 + nt * 16 + nIdx;
        const int h = n >> 6, d = n & (DHEAD - 1);
        const _Float16 val = (_Float16)acc[mi][nt][i];
        if (which == 0)
          q16[(((b << 4) + h) * SEQ + s) * DHEAD + d] = val;
        else if (which == 1)
          k16[(((b << 4) + h) * SEQ + s) * DHEAD + d] = val;
        else
          vT16[(((b << 4) + h) * DHEAD + d) * SEQ + s] = val;
      }
    }
  }
}

// ---------------------------------------------------------------------------
// Phase 2: fused QEr + causal flash attention with TDM K/V streaming.
// grid 1024 = (B*H) x (S/64) ; block 128 (4 waves, 16 query rows each).
// LDS layout (f16 units):
//   [0      , 65536) : per-wave QEr strips  (4 x 16 x 1024)
//   [65536  , 67584) : per-wave P strips    (4 x 16 x 32)
//   [67584  , 71680) : K stage, 2 bufs of 32x64
//   [71680  , 75776) : V stage, 2 bufs of 64x32
// ---------------------------------------------------------------------------
#define QER_OFF(w)  ((w) * 16384)
#define PST_OFF(w)  (65536 + (w) * 512)
#define KSTG_OFF(bf) (67584 + (bf) * 2048)
#define VSTG_OFF(bf) (71680 + (bf) * 2048)
#define SMEM_HALFS   75776

__global__ void __launch_bounds__(128)
rga_attn_kernel(const _Float16* __restrict__ q16,
                const _Float16* __restrict__ k16,
                const _Float16* __restrict__ vT16,
                const _Float16* __restrict__ Er16,
                float* __restrict__ out) {
  extern __shared__ _Float16 smem[];
  const int wave = threadIdx.x >> 5;
  const int lane = lane_id();
  const int half = lane >> 4;
  const int nIdx = lane & 15;

  const int qb = blockIdx.x & 15;  // 64-row query block
  const int bh = blockIdx.x >> 4;  // b*16 + h
  const int b = bh >> 4;
  const int h = bh & 15;

  const _Float16* qh = q16 + (size_t)bh * SEQ * DHEAD;
  const _Float16* kh = k16 + (size_t)bh * SEQ * DHEAD;
  const _Float16* vh = vT16 + (size_t)bh * DHEAD * SEQ;

  const int s0 = qb * 64 + wave * 16;     // wave's first query row
  const int kbmax = qb * 2 + 1;           // uniform across block (64 rows)

  // Kick off TDM for key-block 0 immediately; it overlaps the QEr phase.
  if (wave == 0) {
    tdm_load_2d(KSTG_OFF(0) * 2, kh, 2048, 1, 2048);        // K rows 0..31
    tdm_load_2d(VSTG_OFF(0) * 2, vh, 32, 64, SEQ);          // V^T cols 0..31
  }

  // Q fragments (reused for QEr and all key tiles)
  const v16h qf0 = frag_a_f16(qh + s0 * DHEAD, DHEAD);
  const v16h qf1 = frag_a_f16(qh + s0 * DHEAD + 32, DHEAD);

  // ---- Phase A: QEr[16 x 1024] strip -> per-wave private LDS (f16) ----
  _Float16* qer = smem + QER_OFF(wave);
  for (int rt = 0; rt < 64; ++rt) {
    const int r0 = rt * 16;
    if (rt < 63)  // CDNA5 global_prefetch_b8: next Er tile
      __builtin_prefetch((const void*)(Er16 + (r0 + 16) * DHEAD + lane * 32),
                         0, 1);
    v8f a = {};
    a = wmma_f16(qf0, frag_b_f16(Er16 + r0 * DHEAD, DHEAD), a);
    a = wmma_f16(qf1, frag_b_f16(Er16 + r0 * DHEAD + 32, DHEAD), a);
#pragma unroll
    for (int i = 0; i < 8; ++i) {
      const int mrow = i + (half << 3);
      qer[mrow * 1024 + r0 + nIdx] = (_Float16)a[i];
    }
  }

  _Float16* pst = smem + PST_OFF(wave);  // 16 x 32 P strip

  // ---- Phase B: flash attention over key blocks of 32 ----
  v8f O[4] = {};
  float mrun[8], lrun[8];
#pragma unroll
  for (int i = 0; i < 8; ++i) { mrun[i] = -3.0e38f; lrun[i] = 0.0f; }
  const float scale = 0.125f;  // 1/sqrt(64)

  for (int kb = 0; kb <= kbmax; ++kb) {
    const int t0 = kb * 32;
    const int buf = kb & 1;

    // Barrier A: everyone is done reading buf^1 (iteration kb-1) before the
    // TDM overwrites it; then wave 0 pipelines the next tile + waits for the
    // current one (tensor ops complete in order: <=2 leaves only the newer
    // pair outstanding).
    __syncthreads();
    if (wave == 0) {
      if (kb < kbmax) {
        const int tn = t0 + 32;
        tdm_load_2d(KSTG_OFF(buf ^ 1) * 2, kh + tn * DHEAD, 2048, 1, 2048);
        tdm_load_2d(VSTG_OFF(buf ^ 1) * 2, vh + tn, 32, 64, SEQ);
        __builtin_amdgcn_s_wait_tensorcnt(2);
      } else {
        __builtin_amdgcn_s_wait_tensorcnt(0);
      }
    }
    __syncthreads();  // Barrier B: buf is ready for all waves

    const _Float16* kst = smem + KSTG_OFF(buf);  // [32][64]
    const _Float16* vst = smem + VSTG_OFF(buf);  // [64][32]

    // raw scores: two 16x16 tiles over keys [t0, t0+32)
    v8f a0 = {}, a1 = {};
    a0 = wmma_f16(qf0, frag_b_f16(kst, DHEAD), a0);
    a0 = wmma_f16(qf1, frag_b_f16(kst + 32, DHEAD), a0);
    a1 = wmma_f16(qf0, frag_b_f16(kst + 16 * DHEAD, DHEAD), a1);
    a1 = wmma_f16(qf1, frag_b_f16(kst + 16 * DHEAD + 32, DHEAD), a1);

    float p0[8], p1[8], mnew[8];
#pragma unroll
    for (int i = 0; i < 8; ++i) {
      const int mrow = i + (half << 3);
      const int s = s0 + mrow;
      const int ta = t0 + nIdx;
      const int tb = ta + 16;
      // skew gather: Srel[s,t] = QEr[s, t - s + 1023]
      const float x0 = (ta <= s)
          ? (a0[i] + (float)qer[mrow * 1024 + (ta - s + 1023)]) * scale
          : -3.0e38f;
      const float x1 = (tb <= s)
          ? (a1[i] + (float)qer[mrow * 1024 + (tb - s + 1023)]) * scale
          : -3.0e38f;
      p0[i] = x0;
      p1[i] = x1;
      const float rm = rowmax16(fmaxf(x0, x1));
      mnew[i] = fmaxf(mrun[i], rm);
    }

#pragma unroll
    for (int i = 0; i < 8; ++i) {
      const float e0 = __expf(p0[i] - mnew[i]);  // masked -> 0
      const float e1 = __expf(p1[i] - mnew[i]);
      const float rs = rowsum16(e0 + e1);
      const float alpha = __expf(mrun[i] - mnew[i]);
      lrun[i] = lrun[i] * alpha + rs;
      mrun[i] = mnew[i];
#pragma unroll
      for (int nt = 0; nt < 4; ++nt) O[nt][i] *= alpha;
      const int mrow = i + (half << 3);
      pst[mrow * 32 + nIdx] = (_Float16)e0;       // accumulator -> A layout
      pst[mrow * 32 + 16 + nIdx] = (_Float16)e1;  // via per-wave LDS strip
    }

    const v16h pf = frag_a_f16(pst, 32);  // P tile 16x32 (K = key index)
#pragma unroll
    for (int nt = 0; nt < 4; ++nt) {
      O[nt] = wmma_f16(pf, frag_b_f16(vst + (nt * 16) * 32, 32), O[nt]);
    }
  }

  // ---- epilogue: normalize rows, write out[b, s, h*64 + d] (f32) ----
  float* ob = out + ((size_t)b * SEQ + s0) * DMODEL + h * DHEAD;
#pragma unroll
  for (int i = 0; i < 8; ++i) {
    const float inv = 1.0f / lrun[i];
    const int mrow = i + (half << 3);
#pragma unroll
    for (int nt = 0; nt < 4; ++nt) {
      ob[(size_t)mrow * DMODEL + nt * 16 + nIdx] = O[nt][i] * inv;
    }
  }
}

// ---------------------------------------------------------------------------
extern "C" void kernel_launch(void* const* d_in, const int* in_sizes, int n_in,
                              void* d_out, int out_size, void* d_ws,
                              size_t ws_size, hipStream_t stream) {
  (void)in_sizes; (void)n_in; (void)out_size; (void)ws_size;
  const float* x  = (const float*)d_in[0];  // queries (only input used)
  const float* Wq = (const float*)d_in[3];
  const float* Wk = (const float*)d_in[4];
  const float* Wv = (const float*)d_in[5];
  const float* Er = (const float*)d_in[6];
  float* out = (float*)d_out;

  const size_t headElems = (size_t)BATCH * NHEAD * SEQ * DHEAD;  // 4Mi
  _Float16* q16  = (_Float16*)d_ws;
  _Float16* k16  = q16 + headElems;
  _Float16* vT   = k16 + headElems;
  _Float16* x16  = vT + headElems;                         // 4Mi
  _Float16* Wq16 = x16 + (size_t)BATCH * SEQ * DMODEL;     // 1Mi each
  _Float16* Wk16 = Wq16 + (size_t)DMODEL * DMODEL;
  _Float16* Wv16 = Wk16 + (size_t)DMODEL * DMODEL;
  _Float16* Er16 = Wv16 + (size_t)DMODEL * DMODEL;         // 64Ki

  // Phase 0: pre-convert everything the matmuls touch to f16.
  const int n8x = (BATCH * SEQ * DMODEL) / 8;
  cvt_f16_kernel<<<dim3((n8x + 255) / 256), 256, 0, stream>>>(x, x16, n8x);
  const int n8w = (DMODEL * DMODEL) / 8;
  cvt_f16_kernel<<<dim3((n8w + 255) / 256), 256, 0, stream>>>(Wq, Wq16, n8w);
  cvt_f16_kernel<<<dim3((n8w + 255) / 256), 256, 0, stream>>>(Wk, Wk16, n8w);
  cvt_f16_kernel<<<dim3((n8w + 255) / 256), 256, 0, stream>>>(Wv, Wv16, n8w);
  const int n8e = (SEQ * DHEAD) / 8;
  cvt_f16_kernel<<<dim3((n8e + 255) / 256), 256, 0, stream>>>(Er, Er16, n8e);

  // Phase 1: QKV projections.
  qkv_proj_kernel<<<dim3(16, 16, 3), 256, 0, stream>>>(x16, Wq16, Wk16, Wv16,
                                                       q16, k16, vT);

  // Phase 2: fused relative-position flash attention.
  rga_attn_kernel<<<dim3((BATCH * NHEAD) * (SEQ / 64)), 128, SMEM_HALFS * 2,
                    stream>>>(q16, k16, vT, Er16, out);
}
